// TemporalAttention_28973849379424
// MI455X (gfx1250) — matrix operand order repaired and checked by reading, not compiled, not run on hardware
//
#include <hip/hip_runtime.h>

#define H     1024
#define SEQ   2048
#define BATCH 8
#define TOKS  (BATCH * SEQ)      // 16384
#define EPS_LN 1e-5f
#define QSCALE 0.03125f          // 1/sqrt(1024)

typedef __attribute__((ext_vector_type(16))) __bf16       bf16x16;
typedef __attribute__((ext_vector_type(4)))  __bf16       bf16x4;
typedef __attribute__((ext_vector_type(8)))  float        f32x8;
typedef __attribute__((ext_vector_type(4)))  float        f32x4;
typedef __attribute__((ext_vector_type(4)))  unsigned int u32x4;
typedef __attribute__((ext_vector_type(4)))  int          i32x4v;

// trivially-copyable fragment: two 16B chunks == one 16x32 bf16 A/B operand per lane
union Frag {
    u32x4   u[2];
    bf16x16 v;
};

__device__ __forceinline__ f32x8 wmma_bf16(const Frag& a, const Frag& b, f32x8 c) {
    // D = A(16x32 bf16) * B(32x16 bf16) + C(16x16 f32)
    return __builtin_amdgcn_wmma_f32_16x16x32_bf16(false, a.v, false, b.v,
                                                   (short)0, c, false, false);
}

// ---------------- async global->LDS path (guarded; falls back to ld+ds_store) ---
#if defined(__has_builtin)
#  if __has_builtin(__builtin_amdgcn_global_load_async_to_lds_b128) && \
      __has_builtin(__builtin_amdgcn_s_wait_asynccnt)
#    define USE_ASYNC_LDS 1
#  endif
#endif
#ifndef USE_ASYNC_LDS
#  define USE_ASYNC_LDS 0
#endif

#if USE_ASYNC_LDS
// builtin signature (per hipcc diagnostic): (int4 AS1* src, int4 AS3* dst, Ii, Ii)
typedef __attribute__((address_space(1))) i32x4v gvec4;
typedef __attribute__((address_space(3))) i32x4v lvec4;
#endif

// ---------------------------------------------------------------- fp32 -> bf16 --
__global__ void cvt_f32_bf16(const float* __restrict__ src,
                             __bf16* __restrict__ dst, int n4) {
    int i = blockIdx.x * blockDim.x + threadIdx.x;
    int stride = gridDim.x * blockDim.x;
    for (; i < n4; i += stride) {
        f32x4 f = ((const f32x4*)src)[i];
        bf16x4 o;
        o.x = (__bf16)f.x; o.y = (__bf16)f.y;
        o.z = (__bf16)f.z; o.w = (__bf16)f.w;
        ((bf16x4*)dst)[i] = o;
    }
}

// -------------------------------------------------- fused Q/K/V projection GEMM --
// block = 256 threads (8 waves), tile 128(M) x 64(N), K-step 32, x staged in LDS
// (double buffered, async copy when available). W fragments straight from L2.
// Writes: qb (pre-scaled 1/sqrt(H)), kb row-major, vt TRANSPOSED [H][TOKS].
__global__ void __launch_bounds__(256)
qkv_gemm(const __bf16* __restrict__ xb,
         const __bf16* __restrict__ wq, const __bf16* __restrict__ wk,
         const __bf16* __restrict__ wv,
         const float* __restrict__ bq, const float* __restrict__ bk,
         const float* __restrict__ bv,
         __bf16* __restrict__ qb, __bf16* __restrict__ kb,
         __bf16* __restrict__ vt)
{
    __shared__ u32x4 Atile[2][512];            // 2 x (128 rows x 32 bf16) = 16 KB

    const int tid  = threadIdx.x;
    const int lane = tid & 31;
    const int w    = tid >> 5;
    const int wm   = w >> 1, wn = w & 1;       // 4x2 wave grid
    const int mblk = blockIdx.y * 128;
    const int nblk = blockIdx.x * 64;
    const int n15  = lane & 15;
    const int c0   = (lane < 16) ? 0 : 1;      // A-frag chunk select (u32x4 units)
    const int bk16 = (lane < 16) ? 0 : 16;     // B-frag K offset
    const int mh   = (lane < 16) ? 0 : 8;      // C row half

    const f32x8 z = {0.f,0.f,0.f,0.f,0.f,0.f,0.f,0.f};
    f32x8 acc[3][2][2];
    #pragma unroll
    for (int s = 0; s < 3; ++s)
        #pragma unroll
        for (int a = 0; a < 2; ++a)
            #pragma unroll
            for (int b = 0; b < 2; ++b) acc[s][a][b] = z;

    auto load_tile = [&](int buf, int kk) {
        const int k0 = kk * 32;
        #pragma unroll
        for (int t = 0; t < 2; ++t) {
            int c   = tid + t * 256;           // 512 chunks of 16B
            int row = c >> 2, cq = c & 3;
            const __bf16* gsrc = xb + (size_t)(mblk + row) * H + k0 + cq * 8;
#if USE_ASYNC_LDS
            __builtin_amdgcn_global_load_async_to_lds_b128(
                (gvec4*)gsrc, (lvec4*)&Atile[buf][c], 0, 0);
#else
            Atile[buf][c] = *(const u32x4*)gsrc;
#endif
        }
    };

    load_tile(0, 0);
    #pragma unroll 1
    for (int kk = 0; kk < H / 32; ++kk) {
        const int cur = kk & 1;
        if (kk + 1 < H / 32) load_tile(1 - cur, kk + 1);
#if USE_ASYNC_LDS
        if (kk + 1 < H / 32) __builtin_amdgcn_s_wait_asynccnt(2); // cur tile done
        else                 __builtin_amdgcn_s_wait_asynccnt(0);
#endif
        __syncthreads();

        Frag afr[2];
        const u32x4* A4 = Atile[cur];
        #pragma unroll
        for (int mt = 0; mt < 2; ++mt) {
            int row = wm * 32 + mt * 16 + n15;
            afr[mt].u[0] = A4[row * 4 + 2 * c0];
            afr[mt].u[1] = A4[row * 4 + 2 * c0 + 1];
        }

        const int kb_off = kk * 32 + bk16;
        const __bf16* Wl[3] = {wq, wk, wv};
        #pragma unroll
        for (int s = 0; s < 3; ++s)
            #pragma unroll
            for (int nt = 0; nt < 2; ++nt) {
                int o = nblk + wn * 32 + nt * 16 + n15;
                const u32x4* p = (const u32x4*)(Wl[s] + (size_t)o * H + kb_off);
                Frag bfr; bfr.u[0] = p[0]; bfr.u[1] = p[1];
                #pragma unroll
                for (int mt = 0; mt < 2; ++mt)
                    acc[s][mt][nt] = wmma_bf16(afr[mt], bfr, acc[s][mt][nt]);
            }
        __syncthreads();
    }

    #pragma unroll
    for (int s = 0; s < 3; ++s) {
        const float* bias = (s == 0) ? bq : ((s == 1) ? bk : bv);
        #pragma unroll
        for (int mt = 0; mt < 2; ++mt)
            #pragma unroll
            for (int nt = 0; nt < 2; ++nt) {
                int o = nblk + wn * 32 + nt * 16 + n15;
                float bo = bias[o];
                #pragma unroll
                for (int r = 0; r < 8; ++r) {
                    int t  = mblk + wm * 32 + mt * 16 + r + mh;
                    float v = acc[s][mt][nt][r] + bo;
                    if (s == 0)      qb[(size_t)t * H + o] = (__bf16)(v * QSCALE);
                    else if (s == 1) kb[(size_t)t * H + o] = (__bf16)v;
                    else             vt[(size_t)o * TOKS + t] = (__bf16)v; // transposed
                }
            }
    }
}

// ----------------------------------------------------------- flash attention ----
// 1 block per (batch, 16-query tile); 8 waves each own a 128-wide H slice.
// Online softmax, 32 keys/step; cross-wave score reduction via ds_add_f32.
__global__ void __launch_bounds__(256)
flash_attn(const __bf16* __restrict__ qb, const __bf16* __restrict__ kbm,
           const __bf16* __restrict__ vt, __bf16* __restrict__ ctx)
{
    __shared__ float  sred[16][32];                    // partial scores (f32)
    __shared__ __align__(16) __bf16 pbuf[16][32];      // softmaxed P tile (bf16)
    __shared__ float  rstat[2][16];                    // 0: corr, 1: 1/l

    const int tid  = threadIdx.x;
    const int lane = tid & 31;
    const int w    = tid >> 5;
    const int b    = blockIdx.x >> 7;
    const int qt   = blockIdx.x & 127;
    const int qbase = b * SEQ + qt * 16;
    const int h0   = w * 128;
    const int n15  = lane & 15;
    const int c0   = (lane < 16) ? 0 : 1;
    const int k16  = (lane < 16) ? 0 : 16;
    const int mh   = (lane < 16) ? 0 : 8;

    // Q slice: 16 rows x 128 cols -> 4 A-fragments, kept in registers
    Frag aq[4];
    {
        const u32x4* p = (const u32x4*)(qb + (size_t)(qbase + n15) * H + h0);
        #pragma unroll
        for (int c = 0; c < 4; ++c) {
            aq[c].u[0] = p[c * 4 + 2 * c0];
            aq[c].u[1] = p[c * 4 + 2 * c0 + 1];
        }
    }

    const f32x8 z = {0.f,0.f,0.f,0.f,0.f,0.f,0.f,0.f};
    f32x8 acc[8];
    #pragma unroll
    for (int t = 0; t < 8; ++t) acc[t] = z;

    float mrun = -1e30f, lsum = 0.f;   // meaningful on lanes 0..15 of wave 0

    #pragma unroll 1
    for (int kt = 0; kt < SEQ / 32; ++kt) {
        const int kb0 = kt * 32;

        ((float*)sred)[tid]       = 0.f;
        ((float*)sred)[tid + 256] = 0.f;
        __syncthreads();

        // partial scores over this wave's 128-wide H slice
        #pragma unroll
        for (int half = 0; half < 2; ++half) {
            f32x8 sc = z;
            const size_t krow =
                (size_t)(b * SEQ + kb0 + half * 16 + n15) * H + h0 + k16;
            #pragma unroll
            for (int c = 0; c < 4; ++c) {
                const u32x4* p = (const u32x4*)(kbm + krow + c * 32);
                Frag bfr; bfr.u[0] = p[0]; bfr.u[1] = p[1];
                sc = wmma_bf16(aq[c], bfr, sc);
            }
            #pragma unroll
            for (int r = 0; r < 8; ++r)
                atomicAdd(&sred[r + mh][half * 16 + n15], sc[r]);  // ds_add_f32
        }
        __syncthreads();

        // online softmax update: one lane per query row
        if (tid < 16) {
            const int q = tid;
            float mx = -1e30f;
            for (int j = 0; j < 32; ++j) mx = fmaxf(mx, sred[q][j]);
            float mn   = fmaxf(mrun, mx);
            float corr = __expf(mrun - mn);
            float s = 0.f;
            for (int j = 0; j < 32; ++j) {
                float pj = __expf(sred[q][j] - mn);
                pbuf[q][j] = (__bf16)pj;
                s += pj;
            }
            lsum = lsum * corr + s;
            mrun = mn;
            rstat[0][q] = corr;
        }
        __syncthreads();

        // P fragment shared by all 8 h-tiles of this wave
        Frag pa;
        {
            const u32x4* p4 = (const u32x4*)pbuf;
            pa.u[0] = p4[n15 * 4 + 2 * c0];
            pa.u[1] = p4[n15 * 4 + 2 * c0 + 1];
        }
        float corr8[8];
        #pragma unroll
        for (int r = 0; r < 8; ++r) corr8[r] = rstat[0][r + mh];

        #pragma unroll
        for (int t = 0; t < 8; ++t) {
            #pragma unroll
            for (int r = 0; r < 8; ++r) acc[t][r] *= corr8[r];
            const int hc = h0 + t * 16 + n15;          // V^T row = h column
            const u32x4* p =
                (const u32x4*)(vt + (size_t)hc * TOKS + b * SEQ + kb0 + k16);
            Frag bfr; bfr.u[0] = p[0]; bfr.u[1] = p[1];
            acc[t] = wmma_bf16(pa, bfr, acc[t]);
        }
        __syncthreads();
    }

    if (tid < 16) rstat[1][tid] = 1.f / lsum;
    __syncthreads();
    float inv8[8];
    #pragma unroll
    for (int r = 0; r < 8; ++r) inv8[r] = rstat[1][r + mh];
    #pragma unroll
    for (int t = 0; t < 8; ++t)
        #pragma unroll
        for (int r = 0; r < 8; ++r)
            ctx[(size_t)(qbase + r + mh) * H + h0 + t * 16 + n15] =
                (__bf16)(acc[t][r] * inv8[r]);
}

// --------------------------- out-projection + residual + LayerNorm (fused) ------
__global__ void __launch_bounds__(256)
proj_ln(const __bf16* __restrict__ ctx, const __bf16* __restrict__ wp,
        const float* __restrict__ bp, const float* __restrict__ x,
        const float* __restrict__ gamma, const float* __restrict__ beta,
        float* __restrict__ out)
{
    __shared__ float ybuf[16][H];          // 64 KB
    __shared__ float rsum[16], rsq[16];

    const int tid  = threadIdx.x;
    const int lane = tid & 31;
    const int w    = tid >> 5;
    const int tbase = blockIdx.x * 16;
    const int ob   = w * 128;
    const int n15  = lane & 15;
    const int c0   = (lane < 16) ? 0 : 1;
    const int k16  = (lane < 16) ? 0 : 16;
    const int mh   = (lane < 16) ? 0 : 8;

    const f32x8 z = {0.f,0.f,0.f,0.f,0.f,0.f,0.f,0.f};
    f32x8 acc[8];
    #pragma unroll
    for (int t = 0; t < 8; ++t) acc[t] = z;

    #pragma unroll 1
    for (int kk = 0; kk < H / 32; ++kk) {
        const int k0 = kk * 32;
        Frag a;
        const u32x4* pA = (const u32x4*)(ctx + (size_t)(tbase + n15) * H + k0);
        a.u[0] = pA[2 * c0]; a.u[1] = pA[2 * c0 + 1];
        #pragma unroll
        for (int t = 0; t < 8; ++t) {
            const u32x4* pB =
                (const u32x4*)(wp + (size_t)(ob + t * 16 + n15) * H + k0 + k16);
            Frag bfr; bfr.u[0] = pB[0]; bfr.u[1] = pB[1];
            acc[t] = wmma_bf16(a, bfr, acc[t]);
        }
    }

    if (tid < 16) { rsum[tid] = 0.f; rsq[tid] = 0.f; }
    #pragma unroll
    for (int t = 0; t < 8; ++t) {
        int o = ob + t * 16 + n15;
        float bo = bp[o];
        #pragma unroll
        for (int r = 0; r < 8; ++r) {
            int trow = r + mh;
            float v = acc[t][r] + bo + x[(size_t)(tbase + trow) * H + o];
            ybuf[trow][o] = v;
        }
    }
    __syncthreads();
    {
        const int r = tid & 15, seg = tid >> 4;
        const float* yr = ybuf[r] + seg * 64;
        float s = 0.f, s2 = 0.f;
        for (int j = 0; j < 64; ++j) { float v = yr[j]; s += v; s2 += v * v; }
        atomicAdd(&rsum[r], s);            // ds_add_f32
        atomicAdd(&rsq[r], s2);
    }
    __syncthreads();
    {
        const int r = tid & 15, seg = tid >> 4;
        float mu  = rsum[r] * (1.f / H);
        float var = rsq[r] * (1.f / H) - mu * mu;
        float rs  = rsqrtf(var + EPS_LN);
        const float* yr = ybuf[r] + seg * 64;
        const float* g  = gamma + seg * 64;
        const float* be = beta + seg * 64;
        float* orow = out + (size_t)(tbase + r) * H + seg * 64;
        for (int j = 0; j < 64; ++j)
            orow[j] = (yr[j] - mu) * rs * g[j] + be[j];
    }
}

// ------------------------------------------------------------------- launcher ---
extern "C" void kernel_launch(void* const* d_in, const int* in_sizes, int n_in,
                              void* d_out, int out_size, void* d_ws, size_t ws_size,
                              hipStream_t stream) {
    (void)in_sizes; (void)n_in; (void)out_size; (void)ws_size;
    const float* x     = (const float*)d_in[0];
    const float* Wq    = (const float*)d_in[1];
    const float* bq    = (const float*)d_in[2];
    const float* Wk    = (const float*)d_in[3];
    const float* bk    = (const float*)d_in[4];
    const float* Wv    = (const float*)d_in[5];
    const float* bv    = (const float*)d_in[6];
    const float* Wp    = (const float*)d_in[7];
    const float* bp    = (const float*)d_in[8];
    const float* gamma = (const float*)d_in[9];
    const float* beta  = (const float*)d_in[10];
    float* out = (float*)d_out;

    const size_t TH = (size_t)TOKS * H;       // 16.78M elems
    const size_t HH = (size_t)H * H;          // 1.05M elems
    __bf16* ws  = (__bf16*)d_ws;              // total ~176 MB of bf16 staging
    __bf16* xb  = ws;
    __bf16* qb  = xb + TH;
    __bf16* kbm = qb + TH;
    __bf16* vt  = kbm + TH;                   // V transposed: [H][TOKS]
    __bf16* ctx = vt + TH;
    __bf16* wqb = ctx + TH;
    __bf16* wkb = wqb + HH;
    __bf16* wvb = wkb + HH;
    __bf16* wpb = wvb + HH;

    cvt_f32_bf16<<<2048, 256, 0, stream>>>(x,  xb,  (int)(TH / 4));
    cvt_f32_bf16<<<512,  256, 0, stream>>>(Wq, wqb, (int)(HH / 4));
    cvt_f32_bf16<<<512,  256, 0, stream>>>(Wk, wkb, (int)(HH / 4));
    cvt_f32_bf16<<<512,  256, 0, stream>>>(Wv, wvb, (int)(HH / 4));
    cvt_f32_bf16<<<512,  256, 0, stream>>>(Wp, wpb, (int)(HH / 4));

    qkv_gemm<<<dim3(H / 64, TOKS / 128), 256, 0, stream>>>(
        xb, wqb, wkb, wvb, bq, bk, bv, qb, kbm, vt);

    flash_attn<<<BATCH * (SEQ / 16), 256, 0, stream>>>(qb, kbm, vt, ctx);

    proj_ln<<<TOKS / 16, 256, 0, stream>>>(ctx, wpb, bp, x, gamma, beta, out);
}
